// MoE_21526376088106
// MI455X (gfx1250) — matrix operand order repaired and checked by reading, not compiled, not run on hardware
//
#include <hip/hip_runtime.h>

// ---------------------------------------------------------------------------
// MoE layer for MI455X (gfx1250): bf16 WMMA GEMMs, TDM (tensor_load_to_lds)
// staged weight tiles in LDS, gather/scatter experts.
// ---------------------------------------------------------------------------

typedef __attribute__((ext_vector_type(16))) __bf16 v16bf;
typedef __attribute__((ext_vector_type(8)))  float  v8f;
typedef __attribute__((ext_vector_type(4)))  unsigned int v4u;
typedef __attribute__((ext_vector_type(8)))  int    v8i;

namespace moe {
constexpr int B = 8, N = 2048, D = 768, F = 3072, E = 16, K = 2, R = 16;
constexpr int CAP = 160;
constexpr float THRESH = 0.2f, EPS = 1e-9f, LN_EPS = 1e-6f, SCALE = 2.0f;
constexpr float BAL_COEF = 0.01f, Z_COEF = 0.01f;
constexpr int BN = B * N;
constexpr long long BND = (long long)BN * D;
constexpr long long BNF = (long long)BN * F;
}

// ---- WMMA fragment loaders (NT layout: both operands K-contiguous) --------
// A 16x32 bf16: lane&15 = M row; lane>=16 holds K {8..15, 24..31}.
__device__ __forceinline__ v16bf load_afrag(const __bf16* p) {
    union { v16bf v; uint4 q[2]; } u;
    u.q[0] = *reinterpret_cast<const uint4*>(p);
    u.q[1] = *reinterpret_cast<const uint4*>(p + 16);
    return u.v;
}
__device__ __forceinline__ v16bf load_afrag(const float* p) {
    v16bf v;
#pragma unroll
    for (int i = 0; i < 8; ++i) { v[i] = (__bf16)p[i]; v[i + 8] = (__bf16)p[i + 16]; }
    return v;
}
// B 32x16 bf16: lane&15 = N col; lanes>=16 hold K {8..15, 24..31}.
__device__ __forceinline__ v16bf load_bfrag(const __bf16* p) {
    union { v16bf v; uint4 q[2]; } u;
    u.q[0] = *reinterpret_cast<const uint4*>(p);
    u.q[1] = *reinterpret_cast<const uint4*>(p + 16);
    return u.v;
}

// ---- Tensor Data Mover: 2-D bf16 tile (tile_k x tile_rows) -> LDS ----------
// D# group0/group1 per CDNA5 ISA 8.3/8.4; 2-D tensor => VADDR2/3 = NULL form.
__device__ __forceinline__ unsigned lds_off(const void* p) {
    return (unsigned)(unsigned long long)(uintptr_t)p;   // low 32 bits = LDS byte addr
}
__device__ __forceinline__ void tdm_load_tile_bf16(unsigned lds_addr, const void* gptr,
                                                   unsigned tile_k, unsigned tile_rows,
                                                   unsigned stride_elems) {
    unsigned long long ga = (unsigned long long)(uintptr_t)gptr;
    v4u g0;
    g0[0] = 1u;                                              // count=1 (valid), user mode
    g0[1] = lds_addr;                                        // D#.lds_addr (bytes)
    g0[2] = (unsigned)(ga & 0xffffffffull);                  // global_addr[31:0]
    g0[3] = (unsigned)((ga >> 32) & 0x01ffffffull) | (2u << 30); // [56:32], type=2
    v8i g1;
    g1[0] = 1 << 16;                                         // wg_mask=0, data_size=1 (2B)
    g1[1] = (int)(0xffffu << 16);                            // tensor_dim0[15:0] (huge)
    g1[2] = (int)(0x7fffu | (0xffffu << 16));                // td0[31:16] | td1[15:0]
    g1[3] = (int)(0x7fffu | (tile_k << 16));                 // td1[31:16] | tile_dim0
    g1[4] = (int)tile_rows;                                  // tile_dim1 ; tile_dim2=0
    g1[5] = (int)stride_elems;                               // tensor_dim0_stride[31:0]
    g1[6] = 0;                                               // stride hi / dim1_stride lo
    g1[7] = 0;                                               // dim1_stride hi
    asm volatile("tensor_load_to_lds %0, %1" :: "s"(g0), "s"(g1) : "memory");
}

// ---- Dense NT GEMM with TDM-staged B tiles --------------------------------
// C[M,Nc] = act(A[M,Kd] * Bw[Nc,Kd]^T); B tile (BCOLS x 32) double-buffered
// in LDS via tensor_load_to_lds, compute overlapped with the DMA.
template <typename AT, int WM, int WNW, int WN, bool RELU>
__global__ __launch_bounds__(32 * WM * WNW)
void gemm_nt_tdm(const AT* __restrict__ A, const __bf16* __restrict__ Bw,
                 float* __restrict__ C, int M, int Nc, int Kd) {
    constexpr int BCOLS = 16 * WN * WNW;                 // 128
    __shared__ __bf16 bsh[2][BCOLS * 32];
    (void)M;
    const int lane = threadIdx.x & 31;
    const int w    = threadIdx.x >> 5;
    const int wm = w % WM, wn = w / WM;
    const int r = lane & 15, hi = lane >> 4;
    const int row0   = blockIdx.y * (16 * WM) + wm * 16;
    const int colblk = blockIdx.x * BCOLS;
    const AT* arow = A + (size_t)(row0 + r) * Kd + hi * 8;
    if (w == 0)
        tdm_load_tile_bf16(lds_off(&bsh[0][0]), Bw + (size_t)colblk * Kd,
                           32u, (unsigned)BCOLS, (unsigned)Kd);
    v8f acc[WN] = {};
    for (int k0 = 0; k0 < Kd; k0 += 32) {
        const int cur = (k0 >> 5) & 1;
        if (w == 0) __builtin_amdgcn_s_wait_tensorcnt(0);
        __syncthreads();                                  // cur tile ready; prev buf free
        if (w == 0 && k0 + 32 < Kd)
            tdm_load_tile_bf16(lds_off(&bsh[cur ^ 1][0]),
                               Bw + (size_t)colblk * Kd + (k0 + 32),
                               32u, (unsigned)BCOLS, (unsigned)Kd);
        v16bf a = load_afrag(arow + k0);
        __builtin_prefetch(arow + k0 + 256, 0, 0);        // global_prefetch_b8
        const __bf16* bbase = &bsh[cur][(wn * (16 * WN)) * 32];
#pragma unroll
        for (int i = 0; i < WN; ++i) {
            v16bf bf = load_bfrag(bbase + (i * 16 + r) * 32 + hi * 8);  // ds_load_b128
            acc[i] = __builtin_amdgcn_wmma_f32_16x16x32_bf16(
                false, a, false, bf, (short)0, acc[i], false, false);
        }
    }
#pragma unroll
    for (int i = 0; i < WN; ++i) {
        const int cc = colblk + wn * (16 * WN) + i * 16 + r;
#pragma unroll
        for (int j = 0; j < 8; ++j) {
            const int rr = row0 + j + hi * 8;   // C layout: vgpr j -> M=j / j+8
            float v = acc[i][j];
            if (RELU) v = fmaxf(v, 0.0f);
            C[(size_t)rr * Nc + cc] = v;
        }
    }
}

// ---- Plain NT GEMM (small Nc, e.g. the 16-expert gate) --------------------
template <typename AT, int WM, int WNW, int WN, bool RELU>
__global__ __launch_bounds__(32 * WM * WNW)
void gemm_nt(const AT* __restrict__ A, const __bf16* __restrict__ Bw,
             float* __restrict__ C, int M, int Nc, int Kd) {
    (void)M;
    const int lane = threadIdx.x & 31;
    const int w    = threadIdx.x >> 5;
    const int wm = w % WM, wn = w / WM;
    const int r = lane & 15, hi = lane >> 4;
    const int row0 = blockIdx.y * (16 * WM) + wm * 16;
    const int col0 = blockIdx.x * (16 * WN * WNW) + wn * (16 * WN);
    const AT*     arow  = A  + (size_t)(row0 + r) * Kd + hi * 8;
    const __bf16* bcol0 = Bw + (size_t)(col0 + r) * Kd + hi * 8;
    v8f acc[WN] = {};
    for (int k0 = 0; k0 < Kd; k0 += 32) {
        v16bf a = load_afrag(arow + k0);
        __builtin_prefetch(arow + k0 + 256, 0, 0);
#pragma unroll
        for (int i = 0; i < WN; ++i) {
            v16bf bf = load_bfrag(bcol0 + (size_t)i * 16 * Kd + k0);
            acc[i] = __builtin_amdgcn_wmma_f32_16x16x32_bf16(
                false, a, false, bf, (short)0, acc[i], false, false);
        }
    }
#pragma unroll
    for (int i = 0; i < WN; ++i) {
        const int cc = col0 + i * 16 + r;
#pragma unroll
        for (int j = 0; j < 8; ++j) {
            const int rr = row0 + j + hi * 8;
            float v = acc[i][j];
            if (RELU) v = fmaxf(v, 0.0f);
            C[(size_t)rr * Nc + cc] = v;
        }
    }
}

// ---- Expert GEMM: row gather + TDM-staged weights + LoRA + scatter-add ----
template <typename AT, int WM, int WNW, int WN>
__global__ __launch_bounds__(32 * WM * WNW)
void expert_gemm_scatter(const AT* __restrict__ Atoks,       // [B*N, Kd]
                         const __bf16* __restrict__ Bw,      // [Fdim, Kd]
                         const int* __restrict__ slot_token, // [B*E*CAP]
                         const float* __restrict__ slot_gate,// [B*E*CAP]
                         const float* __restrict__ T,        // [B*E*CAP, R]
                         const float* __restrict__ Lw,       // [E, R, Fdim]
                         float* __restrict__ Cout,           // [B*N, Fdim]
                         int Kd, int Fdim) {
    using namespace moe;
    constexpr int BCOLS = 16 * WN * WNW;                 // 128
    __shared__ __bf16 bsh[2][BCOLS * 32];
    const int z = blockIdx.z, b = z / E, e = z % E;
    const int lane = threadIdx.x & 31;
    const int w    = threadIdx.x >> 5;
    const int wm = w % WM, wn = w / WM;
    const int r = lane & 15, hi = lane >> 4;
    const int row0   = blockIdx.y * (16 * WM) + wm * 16;
    const int colblk = blockIdx.x * BCOLS;
    const int slotbase = z * CAP;

    const int ctok = slot_token[slotbase + row0 + r];
    const AT* arow = (ctok >= 0) ? (Atoks + ((size_t)b * N + ctok) * Kd + hi * 8)
                                 : nullptr;
    if (w == 0)
        tdm_load_tile_bf16(lds_off(&bsh[0][0]), Bw + (size_t)colblk * Kd,
                           32u, (unsigned)BCOLS, (unsigned)Kd);
    v8f acc[WN] = {};
    for (int k0 = 0; k0 < Kd; k0 += 32) {
        const int cur = (k0 >> 5) & 1;
        if (w == 0) __builtin_amdgcn_s_wait_tensorcnt(0);
        __syncthreads();
        if (w == 0 && k0 + 32 < Kd)
            tdm_load_tile_bf16(lds_off(&bsh[cur ^ 1][0]),
                               Bw + (size_t)colblk * Kd + (k0 + 32),
                               32u, (unsigned)BCOLS, (unsigned)Kd);
        v16bf a = {};
        if (arow) a = load_afrag(arow + k0);
        const __bf16* bbase = &bsh[cur][(wn * (16 * WN)) * 32];
#pragma unroll
        for (int i = 0; i < WN; ++i) {
            v16bf bf = load_bfrag(bbase + (i * 16 + r) * 32 + hi * 8);
            acc[i] = __builtin_amdgcn_wmma_f32_16x16x32_bf16(
                false, a, false, bf, (short)0, acc[i], false, false);
        }
    }
    float* Cb = Cout + (size_t)b * N * Fdim;
#pragma unroll
    for (int i = 0; i < WN; ++i) {
        const int cc = colblk + wn * (16 * WN) + i * 16 + r;
        float lcol[R];
#pragma unroll
        for (int q = 0; q < R; ++q) lcol[q] = Lw[((size_t)e * R + q) * Fdim + cc];
#pragma unroll
        for (int j = 0; j < 8; ++j) {
            const int c   = row0 + j + hi * 8;
            const int tok = slot_token[slotbase + c];
            if (tok >= 0) {
                const float g = slot_gate[slotbase + c];
                const float* tr = T + (size_t)(slotbase + c) * R;
                float lora = 0.f;
#pragma unroll
                for (int q = 0; q < R; ++q) lora = fmaf(tr[q], lcol[q], lora);
                atomicAdd(&Cb[(size_t)tok * Fdim + cc], g * (acc[i][j] + SCALE * lora));
            }
        }
    }
}

// ---- RMSNorm (T5 style) -> bf16 tokens ------------------------------------
__global__ void rmsnorm_kernel(const float* __restrict__ x,
                               const float* __restrict__ lnw,
                               __bf16* __restrict__ wi_bf) {
    using namespace moe;
    __shared__ float red[256];
    const int t = blockIdx.x;
    const float* xr = x + (size_t)t * D;
    float s = 0.f;
    for (int d = threadIdx.x; d < D; d += blockDim.x) { float v = xr[d]; s += v * v; }
    red[threadIdx.x] = s;
    __syncthreads();
    for (int off = 128; off > 0; off >>= 1) {
        if ((int)threadIdx.x < off) red[threadIdx.x] += red[threadIdx.x + off];
        __syncthreads();
    }
    const float inv = rsqrtf(red[0] / (float)D + LN_EPS);
    for (int d = threadIdx.x; d < D; d += blockDim.x)
        wi_bf[(size_t)t * D + d] = (__bf16)(xr[d] * inv * lnw[d]);
}

__global__ void f32_to_bf16_kernel(const float* __restrict__ in,
                                   __bf16* __restrict__ out, long long n) {
    long long i  = (long long)blockIdx.x * blockDim.x + threadIdx.x;
    long long st = (long long)gridDim.x * blockDim.x;
    for (; i < n; i += st) out[i] = (__bf16)in[i];
}

// ---- Gate: softmax, top-2, stochastic 2nd expert, energy, aux partials ----
__global__ void gate_softmax_topk_kernel(const float* __restrict__ logits,
                                         const float* __restrict__ route,   // [K,B,N]
                                         float* __restrict__ energy_out,
                                         float* __restrict__ gate_v,
                                         int* __restrict__ gate_i,
                                         int* __restrict__ should,
                                         float* __restrict__ dproxy,
                                         float* __restrict__ zl_acc) {
    using namespace moe;
    const int t = blockIdx.x * blockDim.x + threadIdx.x;
    if (t >= BN) return;
    const float* lg = logits + (size_t)t * E;
    float mx = lg[0];
#pragma unroll
    for (int e = 1; e < E; ++e) mx = fmaxf(mx, lg[e]);
    float p[E], se = 0.f;
#pragma unroll
    for (int e = 0; e < E; ++e) { p[e] = expf(lg[e] - mx); se += p[e]; }
    const float lse = mx + logf(se);
    energy_out[t] = -lse;
    atomicAdd(zl_acc, lse * lse);
    const float inv = 1.f / se;
    const int b = t / N;
    float v0 = -1.f, v1 = -1.f; int i0 = 0, i1 = 0;
#pragma unroll
    for (int e = 0; e < E; ++e) {
        const float pe = p[e] * inv;
        atomicAdd(&dproxy[b * E + e], pe);
        if (pe > v0)      { v1 = v0; i1 = i0; v0 = pe; i0 = e; }
        else if (pe > v1) { v1 = pe; i1 = e; }
    }
    const float den = fmaxf(v0 + v1, EPS);
    const float g0 = v0 / den, g1 = v1 / den;
    gate_i[t] = i0;       gate_v[t] = g0;       should[t] = 1;
    gate_i[BN + t] = i1;  gate_v[BN + t] = g1;
    should[BN + t] = (route[(size_t)BN + t] < g1 * (1.f / THRESH)) ? 1 : 0;
}

// ---- Capacity assignment: exclusive cumsum over n, per (b, expert) --------
__global__ void assign_capacity_kernel(const int* __restrict__ gate_i,
                                       const float* __restrict__ gate_v,
                                       const int* __restrict__ should,
                                       int* __restrict__ slot_token,
                                       float* __restrict__ slot_gate,
                                       float* __restrict__ cnt0,
                                       float* __restrict__ gidx_out) {
    using namespace moe;
    const int b = blockIdx.x;
    const int e = threadIdx.x;
    if (e >= E) return;
    int prev = 0, c0 = 0;
    for (int kk = 0; kk < K; ++kk) {
        int run = 0, kept = 0;
        const int base = kk * BN + b * N;
        for (int n = 0; n < N; ++n) {
            if (gate_i[base + n] != e) continue;
            int keptflag = 0;
            if (should[base + n]) {
                const int pos = run + prev;
                if (pos < CAP) {
                    slot_token[(b * E + e) * CAP + pos] = n;
                    slot_gate [(b * E + e) * CAP + pos] = gate_v[base + n];
                    keptflag = 1; ++kept;
                    if (kk == 0) ++c0;
                }
                ++run;
            }
            gidx_out[base + n] = keptflag ? (float)e : 0.f;  // argmax(mask)
        }
        prev += kept;
    }
    cnt0[b * E + e] = (float)c0;
}

// ---- LoRA in-projection: T[b,e,c,r] = x_row . Aw[e,:,r] -------------------
template <typename AT>
__global__ void lora_proj_kernel(const AT* __restrict__ X,       // [B*N, Kd]
                                 const float* __restrict__ Aw,   // [E, Kd, R]
                                 const int* __restrict__ slot_token,
                                 float* __restrict__ T, int Kd) {
    using namespace moe;
    const int z = blockIdx.x, b = z / E, e = z % E;
    for (int q = threadIdx.x; q < CAP * R; q += blockDim.x) {
        const int c = q / R, r = q % R;
        const int tok = slot_token[z * CAP + c];
        float s = 0.f;
        if (tok >= 0) {
            const AT* xr = X + ((size_t)b * N + tok) * Kd;
            const float* aw = Aw + (size_t)e * Kd * R + r;
            for (int d = 0; d < Kd; ++d) s = fmaf((float)xr[d], aw[(size_t)d * R], s);
        }
        T[(size_t)z * CAP * R + q] = s;
    }
}

__global__ void residual_kernel(const float* __restrict__ x,
                                float* __restrict__ out, long long n) {
    long long i  = (long long)blockIdx.x * blockDim.x + threadIdx.x;
    long long st = (long long)gridDim.x * blockDim.x;
    for (; i < n; i += st) out[i] = x[i] + out[i];
}

__global__ void finalize_kernel(const float* __restrict__ dproxy,
                                const float* __restrict__ cnt0,
                                const float* __restrict__ zl_acc,
                                float* __restrict__ outp) {
    using namespace moe;
    if (threadIdx.x == 0 && blockIdx.x == 0) {
        float bal = 0.f;
        for (int i = 0; i < B * E; ++i)
            bal += (dproxy[i] / (float)N) * (cnt0[i] / (float)N);
        bal = bal / (float)(B * E) * (float)(E * E);
        const float zl = zl_acc[0] / (float)(B * N);
        outp[0] = BAL_COEF * bal + Z_COEF * zl;  // total_aux
        outp[1] = bal;
        outp[2] = zl;
    }
}

// ---------------------------------------------------------------------------
extern "C" void kernel_launch(void* const* d_in, const int* in_sizes, int n_in,
                              void* d_out_v, int out_size, void* d_ws, size_t ws_size,
                              hipStream_t stream) {
    using namespace moe;
    (void)in_sizes; (void)n_in; (void)out_size; (void)ws_size;
    const float* x     = (const float*)d_in[0];
    const float* route = (const float*)d_in[1];
    const float* lnw   = (const float*)d_in[2];
    const float* Wg    = (const float*)d_in[3];
    const float* Wi    = (const float*)d_in[4];
    const float* Wo    = (const float*)d_in[5];
    const float* Ai    = (const float*)d_in[6];
    const float* Bi    = (const float*)d_in[7];
    const float* Ao    = (const float*)d_in[8];
    const float* Bo    = (const float*)d_in[9];
    float* out = (float*)d_out_v;

    const long long OFF_TAUX   = BND;
    const long long OFF_LOGITS = BND + 3;
    const long long OFF_ENERGY = OFF_LOGITS + (long long)BN * E;
    const long long OFF_GIDX   = OFF_ENERGY + BN;

    char* wp = (char*)d_ws;
    auto alloc = [&](size_t bytes) -> void* {
        void* p = (void*)wp;
        wp += (bytes + 255) & ~((size_t)255);
        return p;
    };
    __bf16* wi_bf      = (__bf16*)alloc((size_t)BND * 2);          // RMSNorm tokens, bf16
    float*  h          = (float*) alloc((size_t)BNF * 4);          // FFN hidden, f32
    __bf16* Wg_bf      = (__bf16*)alloc((size_t)E * D * 2);
    __bf16* Wi_bf      = (__bf16*)alloc((size_t)F * D * 2);
    __bf16* Wo_bf      = (__bf16*)alloc((size_t)D * F * 2);
    int*    gate_i     = (int*)   alloc((size_t)K * BN * 4);
    float*  gate_v     = (float*) alloc((size_t)K * BN * 4);
    int*    shoulda    = (int*)   alloc((size_t)K * BN * 4);
    int*    slot_token = (int*)   alloc((size_t)B * E * CAP * 4);
    float*  slot_gate  = (float*) alloc((size_t)B * E * CAP * 4);
    float*  t1         = (float*) alloc((size_t)B * E * CAP * R * 4);
    float*  t2         = (float*) alloc((size_t)B * E * CAP * R * 4);
    float*  dproxy     = (float*) alloc((size_t)B * E * 4);
    float*  cnt0       = (float*) alloc((size_t)B * E * 4);
    float*  zl_acc     = (float*) alloc(256);

    hipMemsetAsync(slot_token, 0xFF, (size_t)B * E * CAP * 4, stream); // -1
    hipMemsetAsync(slot_gate,  0,    (size_t)B * E * CAP * 4, stream);
    hipMemsetAsync(dproxy,     0,    (size_t)B * E * 4, stream);
    hipMemsetAsync(cnt0,       0,    (size_t)B * E * 4, stream);
    hipMemsetAsync(zl_acc,     0,    256, stream);

    // 1) RMSNorm -> bf16 tokens
    rmsnorm_kernel<<<BN, 256, 0, stream>>>(x, lnw, wi_bf);
    // 2) weight conversions
    f32_to_bf16_kernel<<<64,   256, 0, stream>>>(Wg, Wg_bf, (long long)E * D);
    f32_to_bf16_kernel<<<1024, 256, 0, stream>>>(Wi, Wi_bf, (long long)F * D);
    f32_to_bf16_kernel<<<1024, 256, 0, stream>>>(Wo, Wo_bf, (long long)D * F);
    // 3) gate logits (WMMA, Nc=16) straight into d_out
    float* logits = out + OFF_LOGITS;
    gemm_nt<__bf16, 8, 1, 1, false>
        <<<dim3(1, BN / 128), 256, 0, stream>>>(wi_bf, Wg_bf, logits, BN, E, D);
    // 4) softmax / top-2 / energy / aux partials
    gate_softmax_topk_kernel<<<BN / 256, 256, 0, stream>>>(
        logits, route, out + OFF_ENERGY, gate_v, gate_i, shoulda, dproxy, zl_acc);
    // 5) capacity assignment (+ gidx)
    assign_capacity_kernel<<<B, E, 0, stream>>>(
        gate_i, gate_v, shoulda, slot_token, slot_gate, cnt0, out + OFF_GIDX);
    // 6) dense h = relu(wi_in @ Wi^T)           [16384x3072x768], TDM-staged B
    gemm_nt_tdm<__bf16, 4, 2, 4, true>
        <<<dim3(F / 128, BN / 64), 256, 0, stream>>>(wi_bf, Wi_bf, h, BN, F, D);
    // 7) LoRA-in projections t1 = xe @ Ai
    lora_proj_kernel<__bf16><<<B * E, 256, 0, stream>>>(wi_bf, Ai, slot_token, t1, D);
    // 8) expert wi (+ LoRA) scatter-add into h
    expert_gemm_scatter<__bf16, 2, 4, 2>
        <<<dim3(F / 128, CAP / 32, B * E), 256, 0, stream>>>(
            wi_bf, Wi_bf, slot_token, slot_gate, t1, Bi, h, D, F);
    // 9) dense o = h @ Wo^T (into d_out)        [16384x768x3072], TDM-staged B
    gemm_nt_tdm<float, 4, 2, 4, false>
        <<<dim3(D / 128, BN / 64), 256, 0, stream>>>(h, Wo_bf, out, BN, D, F);
    // 10) LoRA-in projections t2 = xe2 @ Ao
    lora_proj_kernel<float><<<B * E, 256, 0, stream>>>(h, Ao, slot_token, t2, F);
    // 11) expert wo (+ LoRA) scatter-add into o
    expert_gemm_scatter<float, 2, 4, 2>
        <<<dim3(D / 128, CAP / 32, B * E), 256, 0, stream>>>(
            h, Wo_bf, slot_token, slot_gate, t2, Bo, out, F, D);
    // 12) residual: out = x + o
    residual_kernel<<<2048, 256, 0, stream>>>(x, out, BND);
    // 13) aux scalars
    finalize_kernel<<<1, 32, 0, stream>>>(dproxy, cnt0, zl_acc, out + OFF_TAUX);
}
